// Model_85461259256122
// MI455X (gfx1250) — matrix operand order repaired and checked by reading, not compiled, not run on hardware
//
#include <hip/hip_runtime.h>
#include <hip/hip_bf16.h>
#include <math.h>

// ---------------------------------------------------------------------------
// Hetero-GAT link predictor, restructured for MI455X (gfx1250, wave32).
//
// Live computation (z_artwork in the reference is dead code for the output):
//   z_style = GATConv(x_artwork -> x_style)   (N_STY = 1000 dst nodes)
//   out[l]  = sigmoid( x_artwork[lab_src[l]] . headW_top
//                    + z_style[lab_dst[l]]   . headW_bot + headb )
//
// Restructuring:
//   e_src[i] = x_artwork[i] . (Wsrc @ att_src)       (fold W into vector)
//   e_dst[j] = x_style[j]   . (Wdst @ att_dst)
//   segment max via atomicMax on monotone uint keys (+ fused edge counting)
//   CSR-by-dst build (scan + cursor scatter)  -> no 100M-atomic scatter
//   agg[d][t] = sum_e p_e * x[src_e][t]        (block per dst, LDS staging,
//                                              coalesced 1KB row reads)
//   z_style  = relu( (agg/denom) @ Wsrc + bias )   <-- v_wmma_f32_16x16x32_f16
//   s_art[i] = x_artwork[i] . headW[0:256]          (fused with e_src pass)
//   s_sty[j] = z_style[j]   . headW[256:512]
// ---------------------------------------------------------------------------

#define N_ART 200000
#define N_STY 1000
#define D     256
#define NEDGE 400000
#define NLAB  500000

typedef __attribute__((ext_vector_type(16))) _Float16 v16h;
typedef __attribute__((ext_vector_type(8)))  float    v8f;

__device__ __forceinline__ float wave_sum(float v) {
#pragma unroll
    for (int off = 16; off > 0; off >>= 1) v += __shfl_xor(v, off, 32);
    return v;
}

// float -> monotone unsigned key (and back) for atomicMax-based segment max
__device__ __forceinline__ unsigned f2key(float f) {
    int b = __float_as_int(f);
    return (unsigned)(b ^ ((b >> 31) | 0x80000000));
}
__device__ __forceinline__ float key2f(unsigned k) {
    int b = (k & 0x80000000u) ? (int)(k ^ 0x80000000u) : (int)(~k);
    return __int_as_float(b);
}

// ---- kernel 1: fold attention vectors: v = W @ a  (one wave per row) ------
__global__ void k_fold(const float* __restrict__ Ws, const float* __restrict__ as,
                       const float* __restrict__ Wd, const float* __restrict__ ad,
                       float* __restrict__ vsrc, float* __restrict__ vdst) {
    int wave = (blockIdx.x * blockDim.x + threadIdx.x) >> 5;
    int lane = threadIdx.x & 31;
    if (wave >= 2 * D) return;
    const float* W = (wave < D) ? Ws : Wd;
    const float* a = (wave < D) ? as : ad;
    int r = wave & (D - 1);
    float p = 0.f;
#pragma unroll
    for (int j = lane; j < D; j += 32) p += W[r * D + j] * a[j];
    p = wave_sum(p);
    if (lane == 0) ((wave < D) ? vsrc : vdst)[r] = p;
}

// ---- kernel 2: zero-init scratch region -----------------------------------
__global__ void k_zero(float* __restrict__ p, int n) {
    int i = blockIdx.x * blockDim.x + threadIdx.x;
    if (i < n) p[i] = 0.f;
}

// ---- kernel 3: per-artwork fused dots: e_src and s_art (wave per row) -----
__global__ void k_art_dots(const float* __restrict__ x, const float* __restrict__ vsrc,
                           const float* __restrict__ hwtop,
                           float* __restrict__ e_src, float* __restrict__ s_art) {
    int wave = (blockIdx.x * blockDim.x + threadIdx.x) >> 5;
    int lane = threadIdx.x & 31;
    if (wave >= N_ART) return;
    const float* row = x + (size_t)wave * D + lane * 8;
    float4 r0 = *(const float4*)(row);
    float4 r1 = *(const float4*)(row + 4);
    const float* v = vsrc + lane * 8;
    const float* h = hwtop + lane * 8;
    float d0 = r0.x*v[0] + r0.y*v[1] + r0.z*v[2] + r0.w*v[3]
             + r1.x*v[4] + r1.y*v[5] + r1.z*v[6] + r1.w*v[7];
    float d1 = r0.x*h[0] + r0.y*h[1] + r0.z*h[2] + r0.w*h[3]
             + r1.x*h[4] + r1.y*h[5] + r1.z*h[6] + r1.w*h[7];
    d0 = wave_sum(d0);
    d1 = wave_sum(d1);
    if (lane == 0) { e_src[wave] = d0; s_art[wave] = d1; }
}

// ---- kernel 4: per-style dot (wave per row) -------------------------------
__global__ void k_row_dot(const float* __restrict__ x, const float* __restrict__ v,
                          float* __restrict__ out, int nrows) {
    int wave = (blockIdx.x * blockDim.x + threadIdx.x) >> 5;
    int lane = threadIdx.x & 31;
    if (wave >= nrows) return;
    const float* row = x + (size_t)wave * D + lane * 8;
    const float* vv  = v + lane * 8;
    float d = 0.f;
#pragma unroll
    for (int i = 0; i < 8; ++i) d += row[i] * vv[i];
    d = wave_sum(d);
    if (lane == 0) out[wave] = d;
}

// ---- kernel 5: edge pass 1 — segment max + per-dst edge counting ----------
__global__ void k_edge_max(const int* __restrict__ src, const int* __restrict__ dst,
                           const float* __restrict__ e_src, const float* __restrict__ e_dst,
                           unsigned* __restrict__ mkey, unsigned* __restrict__ count) {
    int e = blockIdx.x * blockDim.x + threadIdx.x;
    if (e >= NEDGE) return;
    int d = dst[e];
    float v = e_src[src[e]] + e_dst[d];
    v = (v > 0.f) ? v : 0.2f * v;               // leaky_relu(0.2)
    atomicMax(&mkey[d], f2key(v));
    atomicAdd(&count[d], 1u);
}

// ---- kernel 6: exclusive scan of counts (single block, Hillis-Steele) -----
__global__ __launch_bounds__(1024) void k_scan(const unsigned* __restrict__ count,
                                               unsigned* __restrict__ offs,
                                               unsigned* __restrict__ cursor) {
    __shared__ unsigned buf[1024];
    int t = threadIdx.x;
    unsigned v = (t < N_STY) ? count[t] : 0u;
    buf[t] = v;
    __syncthreads();
#pragma unroll
    for (int dd = 1; dd < 1024; dd <<= 1) {
        unsigned x = (t >= dd) ? buf[t - dd] : 0u;
        __syncthreads();
        buf[t] += x;
        __syncthreads();
    }
    if (t < N_STY) {
        unsigned excl = buf[t] - v;
        offs[t] = excl;
        cursor[t] = excl;
    }
}

// ---- kernel 7: edge pass 2 — p=exp(e-m); denom += p; CSR scatter ----------
__global__ void k_edge_p(const int* __restrict__ src, const int* __restrict__ dst,
                         const float* __restrict__ e_src, const float* __restrict__ e_dst,
                         const unsigned* __restrict__ mkey,
                         float* __restrict__ denom, unsigned* __restrict__ cursor,
                         float* __restrict__ ps, int* __restrict__ srcs) {
    int e = blockIdx.x * blockDim.x + threadIdx.x;
    if (e >= NEDGE) return;
    int s = src[e], d = dst[e];
    float v = e_src[s] + e_dst[d];
    v = (v > 0.f) ? v : 0.2f * v;
    float p = __expf(v - key2f(mkey[d]));
    atomicAdd(denom + d, p);
    unsigned pos = atomicAdd(cursor + d, 1u);
    ps[pos]   = p;
    srcs[pos] = s;
}

// ---- kernel 8: per-dst aggregation (block per style, LDS-staged edges) ----
// thread t owns dim t: agg[d][t] = sum over edges of p_e * x[src_e][t]
__global__ __launch_bounds__(256) void k_agg(const unsigned* __restrict__ offs,
                                             const unsigned* __restrict__ count,
                                             const float* __restrict__ ps,
                                             const int* __restrict__ srcs,
                                             const float* __restrict__ x,
                                             float* __restrict__ agg) {
    __shared__ float sp[256];
    __shared__ int   ss[256];
    const int d = blockIdx.x;
    const int t = threadIdx.x;
    const unsigned beg = offs[d];
    const unsigned n   = count[d];
    float acc = 0.f;
    for (unsigned base = 0; base < n; base += 256) {
        unsigned chunk = n - base;
        if (chunk > 256) chunk = 256;
        if (t < (int)chunk) {
            sp[t] = ps[beg + base + t];
            ss[t] = srcs[beg + base + t];
        }
        __syncthreads();
        for (unsigned i = 0; i < chunk; ++i) {
            acc += sp[i] * x[(size_t)ss[i] * D + t];   // coalesced 1KB row read
        }
        __syncthreads();
    }
    agg[(size_t)d * D + t] = acc;
}

// ---- kernel 9: z_style = relu((agg/denom) @ W + bias) via WMMA ------------
// One wave per 16x16 output tile. M tiles: 63 (1008 >= 1000), N tiles: 16,
// K loop: 8 steps of 32 (D=256). Layouts per CDNA5 ISA 7.12.2 (16-bit A/B).
__global__ __launch_bounds__(32) void k_gemm_wmma(
        const float* __restrict__ agg, const float* __restrict__ denom,
        const float* __restrict__ W,   const float* __restrict__ bias,
        float* __restrict__ z) {
    const int mt   = blockIdx.x;          // 0..62
    const int nt   = blockIdx.y;          // 0..15
    const int lane = threadIdx.x;         // 0..31
    const int half = lane >> 4;           // which 16-lane group
    const int l16  = lane & 15;
    const int rowA = mt * 16 + l16;       // A-matrix row held by this lane
    const int colB = nt * 16 + l16;       // B-matrix column held by this lane

    float inv = 0.f;
    if (rowA < N_STY) inv = 1.0f / (denom[rowA] + 1e-16f);

    v8f c = {};
#pragma unroll
    for (int kk = 0; kk < 8; ++kk) {
        const int kb = kk * 32;
        // A 16x32 f16: lanes 0-15 hold K=kb+[0..7] (elems 0-7) and kb+[16..23]
        // (elems 8-15); lanes 16-31 hold kb+[8..15] and kb+[24..31].
        v16h a;
        if (rowA < N_STY) {
            const float* ap = agg + (size_t)rowA * D + kb + half * 8;
#pragma unroll
            for (int i = 0; i < 8; ++i) {
                a[i]     = (_Float16)(ap[i]      * inv);
                a[i + 8] = (_Float16)(ap[i + 16] * inv);
            }
        } else {
#pragma unroll
            for (int i = 0; i < 16; ++i) a[i] = (_Float16)0.f;
        }
        // B 32x16 f16: lanes 0-15 hold K=kb+[0..7],[16..23]; lanes 16-31 kb+[8..15],[24..31],
        // column = colB. W is row-major [K=256][N=256].
        v16h b;
        const float* bp = W + (size_t)(kb + half * 8) * D + colB;
#pragma unroll
        for (int i = 0; i < 16; ++i) b[i] = (_Float16)bp[(i + (i >> 3) * 8) * D];

        c = __builtin_amdgcn_wmma_f32_16x16x32_f16(
                false, a, false, b, (short)0, c, false, false);
    }
    // D layout: VGPR r -> M = mt*16 + r + half*8, N = colB
    const float bc = bias[colB];
#pragma unroll
    for (int r = 0; r < 8; ++r) {
        int row = mt * 16 + r + half * 8;
        if (row < N_STY) {
            float v = c[r] + bc;
            z[(size_t)row * D + colB] = (v > 0.f) ? v : 0.f;   // relu
        }
    }
}

// ---- kernel 10: decoder: out = sigmoid(s_art[ls] + s_sty[ld] + b) ---------
__global__ void k_decode(const int* __restrict__ ls, const int* __restrict__ ld,
                         const float* __restrict__ s_art, const float* __restrict__ s_sty,
                         const float* __restrict__ headb, float* __restrict__ out) {
    int l = blockIdx.x * blockDim.x + threadIdx.x;
    if (l >= NLAB) return;
    float v = s_art[ls[l]] + s_sty[ld[l]] + headb[0];
    out[l] = 1.0f / (1.0f + __expf(-v));
}

// ---------------------------------------------------------------------------
extern "C" void kernel_launch(void* const* d_in, const int* in_sizes, int n_in,
                              void* d_out, int out_size, void* d_ws, size_t ws_size,
                              hipStream_t stream) {
    const float* x_art    = (const float*)d_in[0];
    const float* x_sty    = (const float*)d_in[1];
    const int*   src_a2s  = (const int*)  d_in[2];
    const int*   dst_a2s  = (const int*)  d_in[3];
    // d_in[4], d_in[5]: s2a edges — unused (z_artwork is dead in the output)
    const int*   lab_src  = (const int*)  d_in[6];
    const int*   lab_dst  = (const int*)  d_in[7];
    const float* Wsrc     = (const float*)d_in[8];
    const float* Wdst     = (const float*)d_in[9];
    const float* att_src  = (const float*)d_in[10];
    const float* att_dst  = (const float*)d_in[11];
    const float* bias     = (const float*)d_in[12];
    // d_in[13..17]: s2a GAT params — unused
    const float* headW    = (const float*)d_in[18];   // [512,1]
    const float* headb    = (const float*)d_in[19];
    float*       out      = (float*)d_out;

    // workspace layout (32-bit words)
    float* ws      = (float*)d_ws;
    float* v_src   = ws;                  // 256
    float* v_dst   = ws + 256;            // 256
    float* e_dst   = ws + 512;            // 1000
    float* mkeyf   = ws + 1512;           // 1000 (unsigned keys; zero-init)
    float* denom   = ws + 2512;           // 1000 (zero-init)
    float* countf  = ws + 3512;           // 1000 (unsigned; zero-init)
    float* offsf   = ws + 4512;           // 1000 (unsigned)
    float* cursf   = ws + 5512;           // 1000 (unsigned)
    float* s_sty   = ws + 6512;           // 1000
    float* agg     = ws + 7512;           // 1000*256
    float* z_sty   = ws + 263512;         // 1000*256
    float* e_src   = ws + 519512;         // 200000
    float* s_art   = ws + 719512;         // 200000
    float* ps      = ws + 919512;         // 400000 (CSR-sorted p)
    int*   srcs    = (int*)(ws + 1319512);// 400000 (CSR-sorted src ids)
    unsigned* mkey   = (unsigned*)mkeyf;
    unsigned* count  = (unsigned*)countf;
    unsigned* offs   = (unsigned*)offsf;
    unsigned* cursor = (unsigned*)cursf;

    const float* hw_top = headW;          // headW[0:256]
    const float* hw_bot = headW + D;      // headW[256:512]

    // 1) fold attention vectors (512 waves)
    k_fold<<<64, 256, 0, stream>>>(Wsrc, att_src, Wdst, att_dst, v_src, v_dst);

    // 2) zero mkey/denom/count region: ws[1512 .. 4512)
    k_zero<<<(3000 + 255) / 256, 256, 0, stream>>>(mkeyf, 3000);

    // 3) fused per-artwork dots (200000 waves -> 25000 blocks)
    k_art_dots<<<N_ART / 8, 256, 0, stream>>>(x_art, v_src, hw_top, e_src, s_art);

    // 4) per-style attention dot (1000 waves)
    k_row_dot<<<125, 256, 0, stream>>>(x_sty, v_dst, e_dst, N_STY);

    // 5) edge pass 1: segment max + counting
    k_edge_max<<<(NEDGE + 255) / 256, 256, 0, stream>>>(src_a2s, dst_a2s,
                                                        e_src, e_dst, mkey, count);

    // 6) exclusive scan -> CSR offsets + cursors
    k_scan<<<1, 1024, 0, stream>>>(count, offs, cursor);

    // 7) edge pass 2: softmax numerators + denom + CSR scatter of (p, src)
    k_edge_p<<<(NEDGE + 255) / 256, 256, 0, stream>>>(src_a2s, dst_a2s, e_src, e_dst,
                                                      mkey, denom, cursor, ps, srcs);

    // 8) per-dst aggregation (block per style node; coalesced row reads)
    k_agg<<<N_STY, 256, 0, stream>>>(offs, count, ps, srcs, x_art, agg);

    // 9) z_style = relu((agg/denom) @ Wsrc + bias)  — WMMA f16->f32
    dim3 ggrid(63, 16);
    k_gemm_wmma<<<ggrid, 32, 0, stream>>>(agg, denom, Wsrc, bias, z_sty);

    // 10) per-style decoder dot (1000 waves)
    k_row_dot<<<125, 256, 0, stream>>>(z_sty, hw_bot, s_sty, N_STY);

    // 11) decoder gather + sigmoid
    k_decode<<<(NLAB + 255) / 256, 256, 0, stream>>>(lab_src, lab_dst, s_art, s_sty, headb, out);
}
